// MultiHeadAttention_31361851195599
// MI455X (gfx1250) — matrix operand order repaired and checked by reading, not compiled, not run on hardware
//
#include <hip/hip_runtime.h>
#include <cstdint>
#include <cstddef>

// Problem constants (match reference)
#define BATCH   2
#define S_LEN   2048
#define DMODEL  1024
#define NHEADS  16
#define HDIM    64
#define MROWS   (BATCH * S_LEN)   // 4096

#define USE_ASYNC_LDS 1

typedef unsigned short u16;
typedef __attribute__((ext_vector_type(16))) __bf16 v16bf;
typedef __attribute__((ext_vector_type(8)))  float  v8f;

union FragU {
    v16bf v;
    u16   s[16];
    uint4 q[2];
};

// Native f32 -> bf16 convert (lets the compiler use hardware cvt ops)
static __device__ inline u16 f2bf(float f) {
    union { __bf16 b; u16 u; } x;
    x.b = (__bf16)f;
    return x.u;
}

// Issue two async 16B global->LDS copies covering one 32B chunk.
// both the LDS and the global address include the instruction offset,
// so offset:16 advances BOTH sides (ISA 15.18.3 pseudocode).
static __device__ inline void async_copy32(const u16* gsrc, u16* ldst) {
#if USE_ASYNC_LDS
    unsigned lds = (unsigned)(size_t)ldst;          // low 32 bits = LDS offset
    unsigned long long ga = (unsigned long long)(size_t)gsrc;
    asm volatile("global_load_async_to_lds_b128 %0, %1, off\n\t"
                 "global_load_async_to_lds_b128 %0, %1, off offset:16"
                 :: "v"(lds), "v"(ga) : "memory");
#else
    uint4 a0 = ((const uint4*)gsrc)[0];
    uint4 a1 = ((const uint4*)gsrc)[1];
    ((uint4*)ldst)[0] = a0;
    ((uint4*)ldst)[1] = a1;
#endif
}

static __device__ inline void async_wait() {
#if USE_ASYNC_LDS
    asm volatile("s_wait_asynccnt 0" ::: "memory");
#endif
}

// ---------------------------------------------------------------------------
// Pack fp32 -> bf16 (plain row-major)
// ---------------------------------------------------------------------------
__global__ void k_pack_bf16(const float* __restrict__ src, u16* __restrict__ dst, int n) {
    int i = blockIdx.x * blockDim.x + threadIdx.x;
    if (i < n) dst[i] = f2bf(src[i]);
}

// ---------------------------------------------------------------------------
// Pack weight W[K][N] (row-major fp32) into WMMA B-fragment order (bf16),
// optionally pre-scaled (used to fold 1/sqrt(HDIM) into Wq):
//   dst[ ((c*(N/16) + nt)*32 + lane)*16 + i ] = scale * W[ 32c + 16*(lane/16) + i ][ nt*16 + lane%16 ]
// ---------------------------------------------------------------------------
__global__ void k_pack_w(const float* __restrict__ W, u16* __restrict__ Wp, int N, float scale) {
    const int lane = threadIdx.x & 31;
    const int fid  = blockIdx.x * (blockDim.x >> 5) + (threadIdx.x >> 5);
    const int nt   = fid % (N >> 4);
    const int c    = fid / (N >> 4);
    const int nl   = lane & 15, hh = lane >> 4;
    const int n    = nt * 16 + nl;
    u16* out = Wp + ((size_t)fid * 32 + lane) * 16;
#pragma unroll
    for (int i = 0; i < 16; ++i) {
        int k = c * 32 + hh * 16 + i;
        out[i] = f2bf(W[(size_t)k * N + n] * scale);
    }
}

// ---------------------------------------------------------------------------
// GEMM: C[M][N] = A[M][K] (bf16, row-major) @ Bp (bf16, pre-packed B-frags) + bias*bscale
// Block: 256 threads = 8 waves, tile 128(M) x 128(N), K stepped by 32.
// A tile staged through LDS via async global->LDS; B frags straight from packed global (L2).
// ---------------------------------------------------------------------------
template <int OUTF32>
__global__ __launch_bounds__(256) void k_gemm(const u16* __restrict__ A,
                                              const u16* __restrict__ Bp,
                                              const float* __restrict__ bias,
                                              float bscale,
                                              void* __restrict__ Cout,
                                              int M, int N, int K) {
    __shared__ __align__(16) u16 lA[128 * 40];   // 128 rows x 32 elems, pitch 40

    const int tid  = threadIdx.x;
    const int lane = tid & 31, wave = tid >> 5;
    const int nl   = lane & 15, hh = lane >> 4;
    const int wm   = wave >> 2, wn = wave & 3;
    const int mBase = blockIdx.y * 128;
    const int nBase = blockIdx.x * 128;

    v8f acc[4][2];
#pragma unroll
    for (int mi = 0; mi < 4; ++mi)
#pragma unroll
        for (int j = 0; j < 2; ++j)
            acc[mi][j] = (v8f){0.f, 0.f, 0.f, 0.f, 0.f, 0.f, 0.f, 0.f};

    const int KC = K >> 5;
    const int stRow = tid >> 1;
    const int stOff = (tid & 1) * 16;

    for (int kc = 0; kc < KC; ++kc) {
        // Stage A tile (128x32 bf16 = 8KB) to LDS via async DMA, coalesced
        {
            const u16* g = A + (size_t)(mBase + stRow) * K + kc * 32 + stOff;
            async_copy32(g, &lA[stRow * 40 + stOff]);
            if (kc + 1 < KC)
                __builtin_prefetch(g + 32, 0, 0);
        }
        async_wait();
        __syncthreads();

        // B fragments straight from pre-packed global (L2-resident)
        FragU bF[2];
#pragma unroll
        for (int j = 0; j < 2; ++j) {
            int nt = (nBase >> 4) + wn * 2 + j;
            const uint4* g = (const uint4*)(Bp + (((size_t)kc * (N >> 4) + nt) * 32 + lane) * 16);
            bF[j].q[0] = g[0];
            bF[j].q[1] = g[1];
        }

#pragma unroll
        for (int mi = 0; mi < 4; ++mi) {
            FragU aF;
            int row = wm * 64 + mi * 16 + nl;
            aF.q[0] = *(const uint4*)&lA[row * 40 + 16 * hh];
            aF.q[1] = *(const uint4*)&lA[row * 40 + 16 * hh + 8];
#pragma unroll
            for (int j = 0; j < 2; ++j)
                acc[mi][j] = __builtin_amdgcn_wmma_f32_16x16x32_bf16(
                    false, aF.v, false, bF[j].v, (short)0, acc[mi][j], false, false);
        }
        __syncthreads();
    }

    // Epilogue: bias add, store (bf16 or fp32)
#pragma unroll
    for (int mi = 0; mi < 4; ++mi) {
#pragma unroll
        for (int j = 0; j < 2; ++j) {
            int n = nBase + wn * 32 + j * 16 + nl;
            float bv = bias[n] * bscale;
#pragma unroll
            for (int r = 0; r < 8; ++r) {
                int m = mBase + wm * 64 + mi * 16 + r + 8 * hh;
                float v = acc[mi][j][r] + bv;
                if (OUTF32) ((float*)Cout)[(size_t)m * N + n] = v;
                else        ((u16*)Cout)[(size_t)m * N + n]  = f2bf(v);
            }
        }
    }
}

// ---------------------------------------------------------------------------
// Flash attention (causal). Block = (128 queries, 1 head, 1 batch), 128 thr = 4 waves.
// Wave w handles 32 query rows (2 M-fragments) so each K-frag load / V-frag gather
// feeds two WMMAs. K/V streamed in 64-key tiles via async global->LDS.
// All control flow around WMMA uses SCALAR (readfirstlane) conditions so EXEC
// stays all-ones across every v_wmma (ISA 7.12 requirement).
// Q is pre-scaled by 1/sqrt(HDIM) (folded into Wq/bq at pack time).
// ---------------------------------------------------------------------------
__global__ __launch_bounds__(128) void k_attn(const u16* __restrict__ Q,
                                              const u16* __restrict__ Kt,
                                              const u16* __restrict__ Vt,
                                              u16* __restrict__ ctx) {
    __shared__ __align__(16) u16 lK[64 * 72];
    __shared__ __align__(16) u16 lV[64 * 72];
    __shared__ __align__(16) u16 lP[4][2][1024];  // per-wave, per-mq P in A-frag order

    const int tid  = threadIdx.x;
    const int lane = tid & 31, w = tid >> 5;
    const int nl   = lane & 15, hh = lane >> 4;
    const int qb   = blockIdx.x * 128;
    const int head = blockIdx.y, b = blockIdx.z;
    const int qw   = qb + w * 32;
    // Scalar (SGPR) copy of the wave-uniform row base: forces s_cbranch (EXEC
    // stays full) instead of v_cmp + s_and_saveexec around WMMA regions.
    const int qwS  = __builtin_amdgcn_readfirstlane(qw);
    const size_t base = ((size_t)b * S_LEN) * DMODEL + (size_t)head * HDIM;

    // Load Q A-fragments for both 16-row tiles
    FragU qf[2][2];
#pragma unroll
    for (int mq = 0; mq < 2; ++mq) {
        const u16* qp = Q + base + (size_t)(qw + mq * 16 + nl) * DMODEL;
#pragma unroll
        for (int c = 0; c < 2; ++c) {
            qf[mq][c].q[0] = *(const uint4*)(qp + c * 32 + 16 * hh);
            qf[mq][c].q[1] = *(const uint4*)(qp + c * 32 + 16 * hh + 8);
        }
    }

    float mrow[2][8], lrow[2][8];
    v8f oacc[2][4];
#pragma unroll
    for (int mq = 0; mq < 2; ++mq) {
#pragma unroll
        for (int r = 0; r < 8; ++r) { mrow[mq][r] = -1e30f; lrow[mq][r] = 0.f; }
#pragma unroll
        for (int nt = 0; nt < 4; ++nt)
            oacc[mq][nt] = (v8f){0.f, 0.f, 0.f, 0.f, 0.f, 0.f, 0.f, 0.f};
    }

    const int nsteps = 2 * (blockIdx.x + 1);   // causal: keys up to qb+127
    for (int ks = 0; ks < nsteps; ++ks) {
        const int kb0 = ks * 64;
        __syncthreads();   // all waves done reading previous K/V/P
        // Stage K and V 64x64 tiles via async global->LDS (coalesced 128B rows)
#pragma unroll
        for (int it = 0; it < 2; ++it) {
            int chunk = tid + it * 128;            // 256 chunks of 16 elems each
            int row = chunk >> 2, off = (chunk & 3) * 16;
            const u16* gk = Kt + base + (size_t)(kb0 + row) * DMODEL + off;
            const u16* gv = Vt + base + (size_t)(kb0 + row) * DMODEL + off;
            async_copy32(gk, &lK[row * 72 + off]);
            async_copy32(gv, &lV[row * 72 + off]);
            if (ks + 1 < nsteps) {
                __builtin_prefetch(gk + 64 * DMODEL, 0, 0);
                __builtin_prefetch(gv + 64 * DMODEL, 0, 0);
            }
        }
        async_wait();
        __syncthreads();

        // Wave-uniform (scalar) skip of fully-masked steps; kb0 <= qwS when active.
        const bool active = (kb0 <= qwS + 31);

        if (active) {
            // S = Q K^T (pre-scaled), causal mask. D-frag: col = key(nl), rows = r+8*hh
            v8f sf[2][4];
#pragma unroll
            for (int kt = 0; kt < 4; ++kt) {
                // Fragment-level causal classification (all scalar conditions):
                //   fullskip(mq): every element masked -> skip WMMA entirely
                //   needMask(mq): diagonal fragment   -> per-element cndmask
                const int kfirst = kb0 + kt * 16;
                const bool fullskip0 = (kfirst > qwS + 15);
                const bool fullskip1 = (kfirst > qwS + 31);
                const bool needAny   = !fullskip1;   // mq=1 masked => mq=0 masked

                FragU kf[2];
                if (needAny) {
#pragma unroll
                    for (int c = 0; c < 2; ++c) {
                        const u16* p = &lK[(kt * 16 + nl) * 72 + 16 * hh + 32 * c];
                        kf[c].q[0] = ((const uint4*)p)[0];
                        kf[c].q[1] = ((const uint4*)p)[1];
                    }
                }
                int key = kb0 + kt * 16 + nl;
#pragma unroll
                for (int mq = 0; mq < 2; ++mq) {
                    const bool fullskip = mq ? fullskip1 : fullskip0;
                    if (fullskip) {
                        sf[mq][kt] = (v8f){-1e30f, -1e30f, -1e30f, -1e30f,
                                           -1e30f, -1e30f, -1e30f, -1e30f};
                        continue;
                    }
                    v8f s = (v8f){0.f, 0.f, 0.f, 0.f, 0.f, 0.f, 0.f, 0.f};
#pragma unroll
                    for (int c = 0; c < 2; ++c)
                        s = __builtin_amdgcn_wmma_f32_16x16x32_bf16(
                            false, qf[mq][c].v, false, kf[c].v, (short)0, s, false, false);
                    const bool needMask = (kfirst + 15 > qwS + mq * 16);
                    if (needMask) {
#pragma unroll
                        for (int r = 0; r < 8; ++r)
                            if (key > qw + mq * 16 + r + 8 * hh) s[r] = -1e30f;
                    }
                    sf[mq][kt] = s;
                }
            }

            // Online softmax per 16-row tile
#pragma unroll
            for (int mq = 0; mq < 2; ++mq) {
                float tmax[8];
#pragma unroll
                for (int r = 0; r < 8; ++r)
                    tmax[r] = fmaxf(fmaxf(sf[mq][0][r], sf[mq][1][r]),
                                    fmaxf(sf[mq][2][r], sf[mq][3][r]));
#pragma unroll
                for (int off = 1; off < 16; off <<= 1)
#pragma unroll
                    for (int r = 0; r < 8; ++r)
                        tmax[r] = fmaxf(tmax[r], __shfl_xor(tmax[r], off, 32));

                float alpha[8], rsum[8];
#pragma unroll
                for (int r = 0; r < 8; ++r) {
                    float mn = fmaxf(mrow[mq][r], tmax[r]);
                    alpha[r] = __expf(mrow[mq][r] - mn);
                    mrow[mq][r] = mn;
                    rsum[r] = 0.f;
                }
#pragma unroll
                for (int kt = 0; kt < 4; ++kt) {
                    const bool fullskip = (kb0 + kt * 16 > qwS + mq * 16 + 15);
                    if (fullskip) continue;   // exp(-inf)=0 contribution
#pragma unroll
                    for (int r = 0; r < 8; ++r) {
                        float p = __expf(sf[mq][kt][r] - mrow[mq][r]);
                        sf[mq][kt][r] = p;
                        rsum[r] += p;
                    }
                }
#pragma unroll
                for (int off = 1; off < 16; off <<= 1)
#pragma unroll
                    for (int r = 0; r < 8; ++r)
                        rsum[r] += __shfl_xor(rsum[r], off, 32);
#pragma unroll
                for (int r = 0; r < 8; ++r)
                    lrow[mq][r] = lrow[mq][r] * alpha[r] + rsum[r];
#pragma unroll
                for (int nt = 0; nt < 4; ++nt)
#pragma unroll
                    for (int r = 0; r < 8; ++r)
                        oacc[mq][nt][r] *= alpha[r];

                // P (D-frag layout) -> per-wave LDS in A-fragment order
#pragma unroll
                for (int kt = 0; kt < 4; ++kt) {
                    const bool fullskip = (kb0 + kt * 16 > qwS + mq * 16 + 15);
                    if (fullskip) continue;   // not stored; for mq=0 the chunk's O-WMMA may still run
                    int kk    = kt * 16 + nl;
                    int c     = kk >> 5;
                    int khi   = kk & 31;
                    int half2 = (khi >> 3) & 1;
                    int i2    = (khi & 7) + ((khi & 16) ? 8 : 0);
#pragma unroll
                    for (int r = 0; r < 8; ++r) {
                        int lane2 = (r + 8 * hh) + 16 * half2;
                        lP[w][mq][(c * 32 + lane2) * 16 + i2] = f2bf(sf[mq][kt][r]);
                    }
                }
            }
        }
        __syncthreads();

        if (active) {
            // O += P @ V : each V-frag gather feeds up to two WMMAs.
            // K-chunk c is skipped for a tile when both its kt fragments were
            // fully masked (scalar condition).
#pragma unroll
            for (int c = 0; c < 2; ++c) {
                const bool cskip0 = (kb0 + c * 32 > qwS + 15);
                const bool cskip1 = (kb0 + c * 32 > qwS + 31);
                if (cskip0 && cskip1) continue;
                FragU pf[2];
#pragma unroll
                for (int mq = 0; mq < 2; ++mq) {
                    if (mq ? cskip1 : cskip0) continue;
                    const u16* pp = &lP[w][mq][(c * 32 + lane) * 16];
                    pf[mq].q[0] = ((const uint4*)pp)[0];
                    pf[mq].q[1] = ((const uint4*)pp)[1];
                }
#pragma unroll
                for (int nt = 0; nt < 4; ++nt) {
                    FragU vf;   // B-frag of V: column gather from LDS tile
#pragma unroll
                    for (int i = 0; i < 16; ++i)
                        vf.s[i] = lV[(c * 32 + (i < 8 ? 8 * hh + i : 16 + 8 * hh + (i - 8))) * 72 + nt * 16 + nl];
#pragma unroll
                    for (int mq = 0; mq < 2; ++mq) {
                        if (mq ? cskip1 : cskip0) continue;
                        oacc[mq][nt] = __builtin_amdgcn_wmma_f32_16x16x32_bf16(
                            false, pf[mq].v, false, vf.v, (short)0, oacc[mq][nt], false, false);
                    }
                }
            }
        }
    }

    // Finalize: O /= l, store ctx bf16 row-major [B,S,D]
#pragma unroll
    for (int mq = 0; mq < 2; ++mq)
#pragma unroll
        for (int nt = 0; nt < 4; ++nt)
#pragma unroll
            for (int r = 0; r < 8; ++r) {
                int row = qw + mq * 16 + r + 8 * hh;
                float v = oacc[mq][nt][r] / lrow[mq][r];
                ctx[base + (size_t)row * DMODEL + nt * 16 + nl] = f2bf(v);
            }
}

// ---------------------------------------------------------------------------
// Host-side orchestration
// ---------------------------------------------------------------------------
extern "C" void kernel_launch(void* const* d_in, const int* in_sizes, int n_in,
                              void* d_out, int out_size, void* d_ws, size_t ws_size,
                              hipStream_t stream) {
    (void)in_sizes; (void)n_in; (void)out_size; (void)ws_size;

    const float* x  = (const float*)d_in[0];
    const float* Wq = (const float*)d_in[1];
    const float* bq = (const float*)d_in[2];
    const float* Wk = (const float*)d_in[3];
    const float* bk = (const float*)d_in[4];
    const float* Wv = (const float*)d_in[5];
    const float* bv = (const float*)d_in[6];
    const float* Wo = (const float*)d_in[7];
    const float* bo = (const float*)d_in[8];

    // Workspace carve-up (bf16 stored as u16)
    size_t off = 0;
    auto alloc = [&](size_t bytes) -> u16* {
        u16* p = (u16*)((char*)d_ws + off);
        off += (bytes + 255) & ~(size_t)255;
        return p;
    };
    const size_t xdN = (size_t)MROWS * DMODEL;       // 4M elems
    const size_t wN  = (size_t)DMODEL * DMODEL;      // 1M elems
    u16* xb  = alloc(xdN * 2);
    u16* Wqp = alloc(wN * 2);
    u16* Wkp = alloc(wN * 2);
    u16* Wvp = alloc(wN * 2);
    u16* Wop = alloc(wN * 2);
    u16* Qb  = alloc(xdN * 2);
    u16* Kb  = alloc(xdN * 2);
    u16* Vb  = alloc(xdN * 2);
    u16* ctx = alloc(xdN * 2);

    const float qscale = 0.125f;   // 1/sqrt(HDIM), folded into Wq/bq

    // 1) Pack activations and weights to bf16
    {
        int n = (int)xdN;
        k_pack_bf16<<<(n + 255) / 256, 256, 0, stream>>>(x, xb, n);
    }
    {
        dim3 g(2048 / 8);   // (K/32)*(N/16) frags, 8 per 256-thread block
        k_pack_w<<<g, 256, 0, stream>>>(Wq, Wqp, DMODEL, qscale);
        k_pack_w<<<g, 256, 0, stream>>>(Wk, Wkp, DMODEL, 1.0f);
        k_pack_w<<<g, 256, 0, stream>>>(Wv, Wvp, DMODEL, 1.0f);
        k_pack_w<<<g, 256, 0, stream>>>(Wo, Wop, DMODEL, 1.0f);
    }

    // 2) Q/K/V projections (bf16 out); Q pre-scaled
    {
        dim3 g(DMODEL / 128, MROWS / 128);   // (8, 32)
        k_gemm<0><<<g, 256, 0, stream>>>(xb, Wqp, bq, qscale, Qb, MROWS, DMODEL, DMODEL);
        k_gemm<0><<<g, 256, 0, stream>>>(xb, Wkp, bk, 1.0f,  Kb, MROWS, DMODEL, DMODEL);
        k_gemm<0><<<g, 256, 0, stream>>>(xb, Wvp, bv, 1.0f,  Vb, MROWS, DMODEL, DMODEL);
    }

    // 3) Causal flash attention
    {
        dim3 g(S_LEN / 128, NHEADS, BATCH);   // (16, 16, 2)
        k_attn<<<g, 128, 0, stream>>>(Qb, Kb, Vb, ctx);
    }

    // 4) Output projection (fp32 out)
    {
        dim3 g(DMODEL / 128, MROWS / 128);
        k_gemm<1><<<g, 256, 0, stream>>>(ctx, Wop, bo, 1.0f, (float*)d_out, MROWS, DMODEL, DMODEL);
    }
}
